// BaselineEngine_25907242729862
// MI455X (gfx1250) — matrix operand order repaired and checked by reading, not compiled, not run on hardware
//
#include <hip/hip_runtime.h>
#include <hip/hip_bf16.h>

typedef __attribute__((ext_vector_type(16))) _Float16 v16h;
typedef __attribute__((ext_vector_type(8)))  _Float16 v8h;
typedef __attribute__((ext_vector_type(8)))  float    v8f;

#define NCELLS   262144
#define HID      128
#define NFRAG    240           // total weight fragments (each 512 f16 = 1KB)
// fragment table offsets (in fragments)
#define F_WA1    0             // 8 mt * 4 kf   (Wa1[:,64:192]  128x128)
#define F_WG1    32
#define F_WA2    64            // 4 mt * 4 kf   (Wa2 64x128)
#define F_WG2    80
#define F_WIH    96            // 24 mt * 2 kf  (Wih[:,0:64] 384x64)
#define F_WHH    144           // 24 mt * 4 kf  (Whh 384x128)

// ---- d_ws byte layout ----
#define WS_FRAG   0                        // 240*512*2 = 245760 B (f16 fragments)
#define WS_CA     245760                   // 128 f32
#define WS_CG     (WS_CA   + 512)
#define WS_BRZ    (WS_CG   + 512)          // 256 f32  (bih+bhh, rows 0..255)
#define WS_BIHN   (WS_BRZ  + 1024)         // 128 f32  (bih rows 256..383)
#define WS_BHHN   (WS_BIHN + 512)          // 128 f32
#define WS_WT     (WS_BHHN + 512)          // 384 f32  (Wih[:,64])
#define WS_WGPART (WS_WT   + 1536)         // 256*66 f32
#define WS_CHUNK  (WS_WGPART + 67584)      // 1024*128 f32
#define WS_FMEAN  (WS_CHUNK  + 524288)     // 8*128 f32
#define WS_GOP    (WS_FMEAN  + 4096)       // 128 f32

__device__ __forceinline__ v8f wmma16(v16h a, v16h b, v8f c) {
    return __builtin_amdgcn_wmma_f32_16x16x32_f16(false, a, false, b,
                                                  (short)0, c, false, false);
}
__device__ __forceinline__ float sigmf(float x) { return 1.0f / (1.0f + __expf(-x)); }

// ---------------- prep: constants folded from x, bias packs ----------------
__global__ void eng_prep(const float* __restrict__ x,
                         const float* __restrict__ Wa1, const float* __restrict__ ba1,
                         const float* __restrict__ Wg1, const float* __restrict__ bg1,
                         const float* __restrict__ Wih, const float* __restrict__ bih,
                         const float* __restrict__ bhh, unsigned char* ws) {
    int t = threadIdx.x;  // 512 threads, 1 block
    float* ca   = (float*)(ws + WS_CA);
    float* cg   = (float*)(ws + WS_CG);
    float* bRZ  = (float*)(ws + WS_BRZ);
    float* bin  = (float*)(ws + WS_BIHN);
    float* bhn  = (float*)(ws + WS_BHHN);
    float* wt   = (float*)(ws + WS_WT);
    if (t < 128) {
        float sa = ba1[t], sg = bg1[t];
        for (int i = 0; i < 64; ++i) {
            sa += Wa1[t * 192 + i] * x[i];
            sg += Wg1[t * 192 + i] * x[i];
        }
        ca[t] = sa; cg[t] = sg;
        bin[t] = bih[256 + t];
        bhn[t] = bhh[256 + t];
    }
    if (t < 256) bRZ[t] = bih[t] + bhh[t];
    if (t < 384) wt[t] = Wih[t * 65 + 64];
}

// ---------------- pack weights into WMMA A-fragment order (f16) ----------------
__global__ void eng_pack(const float* __restrict__ Wa1, const float* __restrict__ Wg1,
                         const float* __restrict__ Wa2, const float* __restrict__ Wg2,
                         const float* __restrict__ Wih, const float* __restrict__ Whh,
                         unsigned char* ws) {
    int F = blockIdx.x;        // 0..239
    int v = threadIdx.x;       // 0..511
    int lane = v >> 4, i = v & 15;
    const float* W; int ld, coloff, mt, kf;
    if (F < 32)       { int x = F;       W = Wa1; ld = 192; coloff = 64; mt = x >> 2; kf = x & 3; }
    else if (F < 64)  { int x = F - 32;  W = Wg1; ld = 192; coloff = 64; mt = x >> 2; kf = x & 3; }
    else if (F < 80)  { int x = F - 64;  W = Wa2; ld = 128; coloff = 0;  mt = x >> 2; kf = x & 3; }
    else if (F < 96)  { int x = F - 80;  W = Wg2; ld = 128; coloff = 0;  mt = x >> 2; kf = x & 3; }
    else if (F < 144) { int x = F - 96;  W = Wih; ld = 65;  coloff = 0;  mt = x >> 1; kf = x & 1; }
    else              { int x = F - 144; W = Whh; ld = 128; coloff = 0;  mt = x >> 2; kf = x & 3; }
    int m    = mt * 16 + (lane & 15);
    int base = kf * 32 + ((lane >> 4) ? 16 : 0);
    int K    = base + i;
    ((_Float16*)(ws + WS_FRAG))[F * 512 + lane * 16 + i] = (_Float16)W[m * ld + coloff + K];
}

// ---------------- main fused pass: engines + GRU (pre-sync new_h) ----------------
__global__ __launch_bounds__(256) void eng_main(const float* __restrict__ hiddens,
                                                unsigned char* __restrict__ ws,
                                                const float* __restrict__ ba2,
                                                const float* __restrict__ bg2,
                                                float* __restrict__ d_out) {
    __shared__ _Float16 wfrag[NFRAG * 512];      // 245760 B
    __shared__ _Float16 stagebuf[8][2048];       // 32768 B  (B-frag activations, per wave)
    __shared__ _Float16 hstagebuf[8][2048];      // 32768 B  (f16 copy of h, B-frag layout)
    __shared__ float    redbuf[8][66];           // 2112 B

    const int tid    = threadIdx.x;
    const int wave   = tid >> 5;
    const int lane   = tid & 31;
    const int laneM  = lane & 15;
    const int laneHi = lane >> 4;

    { // bulk copy fragments to LDS
        const uint4* src = (const uint4*)(ws + WS_FRAG);
        uint4* dst = (uint4*)wfrag;
        for (int i = tid; i < NFRAG * 512 * 2 / 16; i += 256) dst[i] = src[i];
    }
    __syncthreads();

    _Float16* stage  = stagebuf[wave];
    _Float16* hstage = hstagebuf[wave];
    const float* caP  = (const float*)(ws + WS_CA);
    const float* cgP  = (const float*)(ws + WS_CG);
    const float* bRZ  = (const float*)(ws + WS_BRZ);
    const float* binP = (const float*)(ws + WS_BIHN);
    const float* bhnP = (const float*)(ws + WS_BHHN);
    const float* wtP  = (const float*)(ws + WS_WT);
    float* newh = d_out + 65;

    float cnum[4][8];
    #pragma unroll
    for (int a = 0; a < 4; ++a)
        #pragma unroll
        for (int r = 0; r < 8; ++r) cnum[a][r] = 0.f;
    float cden = 0.f, tsum = 0.f;

    #define AF(F)  (*(const v16h*)&wfrag[(F) * 512 + lane * 16])
    #define BS(kf) (*(const v16h*)&stage[(kf) * 512 + lane * 16])

    for (int it = 0; it < 8; ++it) {
        // keep the compiler from hoisting constant-address LDS fragment loads
        // out of the cell-tile loop (would spill ~2000 VGPRs to scratch)
        asm volatile("" ::: "memory");
        const int cell0 = blockIdx.x * 1024 + it * 128 + wave * 16;
        // ---- load h tile -> B fragments (and f16 copy in LDS) ----
        v16h hB[4];
        const float* hp = hiddens + (size_t)(cell0 + laneM) * 128 + laneHi * 16;
        #pragma unroll
        for (int kf = 0; kf < 4; ++kf) {
            float4 p0 = ((const float4*)(hp + kf * 32))[0];
            float4 p1 = ((const float4*)(hp + kf * 32))[1];
            float4 p2 = ((const float4*)(hp + kf * 32))[2];
            float4 p3 = ((const float4*)(hp + kf * 32))[3];
            v16h hv;
            hv[0]=(_Float16)p0.x; hv[1]=(_Float16)p0.y; hv[2]=(_Float16)p0.z; hv[3]=(_Float16)p0.w;
            hv[4]=(_Float16)p1.x; hv[5]=(_Float16)p1.y; hv[6]=(_Float16)p1.z; hv[7]=(_Float16)p1.w;
            hv[8]=(_Float16)p2.x; hv[9]=(_Float16)p2.y; hv[10]=(_Float16)p2.z; hv[11]=(_Float16)p2.w;
            hv[12]=(_Float16)p3.x; hv[13]=(_Float16)p3.y; hv[14]=(_Float16)p3.z; hv[15]=(_Float16)p3.w;
            hB[kf] = hv;
            *(v16h*)&hstage[kf * 512 + lane * 16] = hv;
        }
        // ---- engine A layer1: relu(WA1h @ h + ca) -> stage (B-frag f16) ----
        #pragma unroll 1
        for (int mt = 0; mt < 8; ++mt) {
            v8f acc = {};
            #pragma unroll
            for (int kf = 0; kf < 4; ++kf) acc = wmma16(AF(F_WA1 + mt * 4 + kf), hB[kf], acc);
            const float* cb = caP + mt * 16 + laneHi * 8;
            v8h o;
            #pragma unroll
            for (int r = 0; r < 8; ++r) {
                float vv = acc[r] + cb[r];
                o[r] = (_Float16)(vv > 0.f ? vv : 0.f);
            }
            *(v8h*)&stage[(mt >> 1) * 512 + (laneM + ((mt & 1) << 4)) * 16 + laneHi * 8] = o;
        }
        // ---- engine A layer2 ----
        v8f outT[4];
        #pragma unroll
        for (int mt2 = 0; mt2 < 4; ++mt2) {
            v8f acc = {};
            #pragma unroll
            for (int kf = 0; kf < 4; ++kf) acc = wmma16(AF(F_WA2 + mt2 * 4 + kf), BS(kf), acc);
            const float* bb = ba2 + mt2 * 16 + laneHi * 8;
            #pragma unroll
            for (int r = 0; r < 8; ++r) acc[r] += bb[r];
            outT[mt2] = acc;
        }
        // ---- engine G layer1 ----
        #pragma unroll 1
        for (int mt = 0; mt < 8; ++mt) {
            v8f acc = {};
            #pragma unroll
            for (int kf = 0; kf < 4; ++kf) acc = wmma16(AF(F_WG1 + mt * 4 + kf), hB[kf], acc);
            const float* cb = cgP + mt * 16 + laneHi * 8;
            v8h o;
            #pragma unroll
            for (int r = 0; r < 8; ++r) {
                float vv = acc[r] + cb[r];
                o[r] = (_Float16)(vv > 0.f ? vv : 0.f);
            }
            *(v8h*)&stage[(mt >> 1) * 512 + (laneM + ((mt & 1) << 4)) * 16 + laneHi * 8] = o;
        }
        // ---- engine G layer2, out = a - g ----
        #pragma unroll
        for (int mt2 = 0; mt2 < 4; ++mt2) {
            v8f acc = {};
            #pragma unroll
            for (int kf = 0; kf < 4; ++kf) acc = wmma16(AF(F_WG2 + mt2 * 4 + kf), BS(kf), acc);
            const float* bb = bg2 + mt2 * 16 + laneHi * 8;
            #pragma unroll
            for (int r = 0; r < 8; ++r) outT[mt2][r] -= acc[r] + bb[r];
        }
        // ---- tension + softmax accumulators ----
        float s = 0.f;
        #pragma unroll
        for (int mt2 = 0; mt2 < 4; ++mt2)
            #pragma unroll
            for (int r = 0; r < 8; ++r) s += outT[mt2][r] * outT[mt2][r];
        s += __shfl_xor(s, 16, 32);
        float tens = s * (1.0f / 64.0f);
        float e = __expf(tens);
        cden += e; tsum += tens;
        #pragma unroll
        for (int mt2 = 0; mt2 < 4; ++mt2)
            #pragma unroll
            for (int r = 0; r < 8; ++r) cnum[mt2][r] += e * outT[mt2][r];
        // ---- stage out (f16 B-frags 0..1) ----
        #pragma unroll
        for (int mt2 = 0; mt2 < 4; ++mt2) {
            v8h o;
            #pragma unroll
            for (int r = 0; r < 8; ++r) o[r] = (_Float16)outT[mt2][r];
            *(v8h*)&stage[(mt2 >> 1) * 512 + (laneM + ((mt2 & 1) << 4)) * 16 + laneHi * 8] = o;
        }
        asm volatile("" ::: "memory");
        // ---- GRU ----
        #pragma unroll 1
        for (int mt = 0; mt < 8; ++mt) {
            v8f gr = {}, gz = {}, gi = {}, gh = {};
            #pragma unroll
            for (int kf = 0; kf < 2; ++kf) gr = wmma16(AF(F_WIH + mt * 2 + kf), BS(kf), gr);
            #pragma unroll
            for (int kf = 0; kf < 4; ++kf) gr = wmma16(AF(F_WHH + mt * 4 + kf), hB[kf], gr);
            #pragma unroll
            for (int kf = 0; kf < 2; ++kf) gz = wmma16(AF(F_WIH + (8 + mt) * 2 + kf), BS(kf), gz);
            #pragma unroll
            for (int kf = 0; kf < 4; ++kf) gz = wmma16(AF(F_WHH + (8 + mt) * 4 + kf), hB[kf], gz);
            #pragma unroll
            for (int kf = 0; kf < 2; ++kf) gi = wmma16(AF(F_WIH + (16 + mt) * 2 + kf), BS(kf), gi);
            #pragma unroll
            for (int kf = 0; kf < 4; ++kf) gh = wmma16(AF(F_WHH + (16 + mt) * 4 + kf), hB[kf], gh);

            const int rowb = mt * 16 + laneHi * 8;
            const float* bR = bRZ + rowb;
            const float* bZ = bRZ + 128 + rowb;
            const float* bN = binP + rowb;
            const float* bH = bhnP + rowb;
            const float* wR = wtP + rowb;
            const float* wZ = wtP + 128 + rowb;
            const float* wN = wtP + 256 + rowb;
            v8h hD = *(const v8h*)&hstage[(mt >> 1) * 512 + (laneM + ((mt & 1) << 4)) * 16 + laneHi * 8];
            float* op = newh + (size_t)(cell0 + laneM) * 128 + mt * 16 + laneHi * 8;
            #pragma unroll
            for (int r = 0; r < 8; ++r) {
                float rg = sigmf(gr[r] + bR[r] + wR[r] * tens);
                float zg = sigmf(gz[r] + bZ[r] + wZ[r] * tens);
                float ng = tanhf(gi[r] + bN[r] + wN[r] * tens + rg * (gh[r] + bH[r]));
                op[r] = (1.f - zg) * ng + zg * (float)hD[r];
            }
        }
    }

    // deterministic lane-group reduction (lanes 0..15 and 16..31 separately)
    #pragma unroll
    for (int m = 1; m < 16; m <<= 1) {
        cden += __shfl_xor(cden, m, 32);
        tsum += __shfl_xor(tsum, m, 32);
        #pragma unroll
        for (int a = 0; a < 4; ++a)
            #pragma unroll
            for (int r = 0; r < 8; ++r) cnum[a][r] += __shfl_xor(cnum[a][r], m, 32);
    }
    if (laneM == 0) {
        #pragma unroll
        for (int a = 0; a < 4; ++a)
            #pragma unroll
            for (int r = 0; r < 8; ++r) redbuf[wave][a * 16 + r + laneHi * 8] = cnum[a][r];
        if (laneHi == 0) { redbuf[wave][64] = cden; redbuf[wave][65] = tsum; }
    }
    __syncthreads();
    if (tid < 66) {
        float acc = 0.f;
        for (int w = 0; w < 8; ++w) acc += redbuf[w][tid];
        ((float*)(ws + WS_WGPART))[blockIdx.x * 66 + tid] = acc;
    }
    #undef AF
    #undef BS
}

// ---------------- faction reductions (deterministic trees) ----------------
__global__ void eng_fpart(const float* __restrict__ newh, float* __restrict__ part) {
    int f = blockIdx.x >> 7, chunk = blockIdx.x & 127, j = threadIdx.x; // 128 thr
    const float* p = newh + (size_t)(f * 32768 + chunk * 256) * 128 + j;
    float s = 0.f;
    for (int c = 0; c < 256; ++c) s += p[(size_t)c * 128];
    part[blockIdx.x * 128 + j] = s;
}
__global__ void eng_fmean(const float* __restrict__ part, float* __restrict__ fmean) {
    int f = blockIdx.x, j = threadIdx.x;
    float s = 0.f;
    for (int ch = 0; ch < 128; ++ch) s += part[(f * 128 + ch) * 128 + j];
    fmean[f * 128 + j] = s * (1.0f / 32768.0f);
}
__global__ void eng_gop(const float* __restrict__ fmean, float* __restrict__ gop) {
    int j = threadIdx.x;
    float s = 0.f;
    for (int f = 0; f < 8; ++f) s += fmean[f * 128 + j];
    gop[j] = s * 0.125f;
}
__global__ __launch_bounds__(256) void eng_sync(float* __restrict__ newh,
                                                const float* __restrict__ fmean,
                                                const float* __restrict__ gop,
                                                const int* __restrict__ stepP) {
    int idx = blockIdx.x * 256 + threadIdx.x;     // one thread per 4 floats
    int e0  = idx * 4;
    int cell = e0 >> 7, j = e0 & 127;
    int f = cell >> 15, cif = cell & 32767;
    bool deb = (*stepP > 5) && (cif < 8192);
    const float* fm = fmean + f * 128 + j;
    const float* gp = gop + j;
    float* p = newh + (size_t)e0;
    #pragma unroll
    for (int u = 0; u < 4; ++u) {
        float v = 0.85f * p[u] + 0.15f * fm[u];
        if (deb) v = 0.85f * v + 0.15f * gp[u];
        p[u] = v;
    }
}
__global__ void eng_final(const float* __restrict__ wgpart, float* __restrict__ d_out) {
    __shared__ float fin[66];
    int j = threadIdx.x; // 128 thr
    if (j < 66) {
        float s = 0.f;
        for (int b = 0; b < 256; ++b) s += wgpart[b * 66 + j];
        fin[j] = s;
    }
    __syncthreads();
    if (j < 64) d_out[j] = fin[j] / fin[64];
    else if (j == 64) d_out[64] = fin[65] * (1.0f / (float)NCELLS);
}

extern "C" void kernel_launch(void* const* d_in, const int* in_sizes, int n_in,
                              void* d_out, int out_size, void* d_ws, size_t ws_size,
                              hipStream_t stream) {
    const float* x    = (const float*)d_in[0];
    const float* hid  = (const float*)d_in[1];
    const float* Wa1  = (const float*)d_in[2];
    const float* ba1  = (const float*)d_in[3];
    const float* ba2  = (const float*)d_in[5];
    const float* bg1  = (const float*)d_in[7];
    const float* bg2  = (const float*)d_in[9];
    const float* Wih  = (const float*)d_in[10];
    const float* bih  = (const float*)d_in[11];
    const float* Whh  = (const float*)d_in[12];
    const float* bhh  = (const float*)d_in[13];
    const int*   step = (const int*)d_in[14];
    const float* Wg1  = (const float*)d_in[6];
    float* out = (float*)d_out;
    unsigned char* ws = (unsigned char*)d_ws;
    (void)in_sizes; (void)n_in; (void)out_size; (void)ws_size;

    eng_prep<<<1, 512, 0, stream>>>(x, Wa1, ba1, Wg1, bg1, Wih, bih, bhh, ws);
    eng_pack<<<NFRAG, 512, 0, stream>>>(Wa1, Wg1, (const float*)d_in[4], (const float*)d_in[8],
                                        Wih, Whh, ws);
    eng_main<<<256, 256, 0, stream>>>(hid, ws, ba2, bg2, out);
    eng_fpart<<<1024, 128, 0, stream>>>(out + 65, (float*)(ws + WS_CHUNK));
    eng_fmean<<<8, 128, 0, stream>>>((float*)(ws + WS_CHUNK), (float*)(ws + WS_FMEAN));
    eng_gop<<<1, 128, 0, stream>>>((float*)(ws + WS_FMEAN), (float*)(ws + WS_GOP));
    eng_sync<<<32768, 256, 0, stream>>>(out + 65, (float*)(ws + WS_FMEAN),
                                        (float*)(ws + WS_GOP), step);
    eng_final<<<1, 128, 0, stream>>>((float*)(ws + WS_WGPART), out);
}
